// intraAFL_Block_3805341024347
// MI455X (gfx1250) — matrix operand order repaired and checked by reading, not compiled, not run on hardware
//
#include <hip/hip_runtime.h>
#include <hip/hip_bf16.h>

// ---------------------------------------------------------------------------
// CDNA5 (gfx1250) implementation: all GEMMs via v_wmma_f32_16x16x32_bf16,
// LDS staging via GLOBAL_LOAD_ASYNC_TO_LDS_B128 (ASYNCcnt path).
// ---------------------------------------------------------------------------

typedef __bf16 bf16_t;
typedef __attribute__((ext_vector_type(16))) __bf16 v16bf;
typedef __attribute__((ext_vector_type(8)))  float  v8f;

#define GF_BIAS  1
#define GF_RELU  2
#define GF_ACCUM 4

__device__ __forceinline__ bf16_t f2bf(float f) {
  union { float f; unsigned u; } x; x.f = f;
  unsigned r = x.u + 0x7FFFu + ((x.u >> 16) & 1u);           // round-nearest-even
  union { unsigned short s; bf16_t b; } y; y.s = (unsigned short)(r >> 16);
  return y.b;
}

union FragAB { v16bf v; uint4 q[2]; };

// A-matrix 16x32 bf16 fragment (ISA 7.12.2): lanes 0-15 row M=lane hold K=0..7,16..23;
// lanes 16-31 same rows hold K=8..15,24..31.  `tile` points at (m0,k0), lda in elems.
__device__ __forceinline__ v16bf load_frag_a(const bf16_t* tile, int lda, int lane) {
  int half = lane >> 4, idx = lane & 15;
  const bf16_t* r = tile + (size_t)idx * lda;
  FragAB f;
  f.q[0] = *(const uint4*)(r + 8 * half);
  f.q[1] = *(const uint4*)(r + 16 + 8 * half);
  return f.v;
}

// B-matrix 32x16 fragment from a row-major [N,K] operand (computes A.B^T):
// lane n (0-15) column N=n holds K=0..7,16..23; lanes 16-31 hold K=8..15,24..31.
__device__ __forceinline__ v16bf load_frag_b(const bf16_t* tile, int ldb, int lane) {
  int half = lane >> 4, idx = lane & 15;
  const bf16_t* r = tile + (size_t)idx * ldb;
  FragAB f;
  f.q[0] = *(const uint4*)(r + 8 * half);
  f.q[1] = *(const uint4*)(r + 16 + 8 * half);
  return f.v;
}

__device__ __forceinline__ v8f wmma_bf16(v16bf a, v16bf b, v8f c) {
  return __builtin_amdgcn_wmma_f32_16x16x32_bf16(false, a, false, b, (short)0, c,
                                                 false, false);
}

__device__ __forceinline__ float wred_sum(float v) {
  #pragma unroll
  for (int o = 16; o > 0; o >>= 1) v += __shfl_xor(v, o, 32);
  return v;
}
__device__ __forceinline__ float wred_max(float v) {
  #pragma unroll
  for (int o = 16; o > 0; o >>= 1) v = fmaxf(v, __shfl_xor(v, o, 32));
  return v;
}

// ---------------------------------------------------------------------------
// Async global->LDS 16B copy (gfx1250 GLOBAL_LOAD_ASYNC_TO_LDS_B128, ASYNCcnt).
// Builtin signature (from toolchain diagnostic): param0 is a non-const
// vector_size(16) int pointer in AS1 ("__device__"); param1 mirrors in AS3.
// Generic global addr == as1 addr; low 32 bits of a generic LDS pointer are the
// wave-relative LDS offset (flat LDS aperture truncation, ISA 10.2).
// ---------------------------------------------------------------------------
#ifndef __has_builtin
#define __has_builtin(x) 0
#endif

typedef int v4i_vs __attribute__((vector_size(16)));
typedef __attribute__((address_space(1))) v4i_vs* as1_v4i;
typedef __attribute__((address_space(3))) v4i_vs* as3_v4i;

__device__ __forceinline__ void async_copy_b128(void* lds_dst, const void* gsrc) {
#if __has_builtin(__builtin_amdgcn_global_load_async_to_lds_b128)
  __builtin_amdgcn_global_load_async_to_lds_b128(
      (as1_v4i)(uintptr_t)gsrc,
      (as3_v4i)(unsigned)(uintptr_t)lds_dst,
      0, 0);
#else
  unsigned lds_off = (unsigned)(uintptr_t)lds_dst;
  asm volatile("global_load_async_to_lds_b128 %0, %1, off"
               :: "v"(lds_off), "v"(gsrc) : "memory");
#endif
}

__device__ __forceinline__ void async_wait0() {
#if __has_builtin(__builtin_amdgcn_s_wait_asynccnt)
  __builtin_amdgcn_s_wait_asynccnt(0);
#else
  asm volatile("s_wait_asynccnt 0x0" ::: "memory");
#endif
}

// ---------------------------------------------------------------------------
// fp32 -> bf16 convert
// ---------------------------------------------------------------------------
__global__ __launch_bounds__(256) void cvt_kernel(const float* __restrict__ in,
                                                  bf16_t* __restrict__ out, int n) {
  int i = blockIdx.x * 256 + threadIdx.x;
  int stride = gridDim.x * 256;
  for (; i < n; i += stride) out[i] = f2bf(in[i]);
}

// ---------------------------------------------------------------------------
// Generic GEMM: C[M,N] = A[M,K] (bf16) . B[N,K]^T (bf16)  (+bias,+accum,+relu)
// 128x64 tile per 256-thread workgroup, 8 waves -> 32x32 each (2x2 WMMA frags).
// ---------------------------------------------------------------------------
#define GBM 128
#define GBN 64
#define GKB 32
#define GLD (GKB + 8)

__global__ __launch_bounds__(256)
void gemm_bf16_kernel(const bf16_t* __restrict__ A, const bf16_t* __restrict__ Bw,
                      const float* __restrict__ bias, float* Cf, bf16_t* Cbf,
                      int M, int N, int K, int flags) {
  __shared__ __align__(16) bf16_t sA[GBM * GLD];
  __shared__ __align__(16) bf16_t sB[GBN * GLD];
  int tid = threadIdx.x, lane = tid & 31, wv = tid >> 5;
  int wm = (wv >> 1) * 32, wn = (wv & 1) * 32;
  int bm = blockIdx.y * GBM, bn = blockIdx.x * GBN;
  const bf16_t* Ab = A + (size_t)bm * K;
  const bf16_t* Bb = Bw + (size_t)bn * K;
  v8f acc[2][2] = {};
  for (int k0 = 0; k0 < K; k0 += GKB) {
    // async stage A (128x32) and B (64x32) tiles into LDS, 16B per lane
    for (int u = tid; u < GBM * 4; u += 256) {
      int r = u >> 2, s4 = u & 3;
      async_copy_b128(sA + r * GLD + s4 * 8, Ab + (size_t)r * K + k0 + s4 * 8);
    }
    for (int u = tid; u < GBN * 4; u += 256) {
      int r = u >> 2, s4 = u & 3;
      async_copy_b128(sB + r * GLD + s4 * 8, Bb + (size_t)r * K + k0 + s4 * 8);
    }
    if (k0 + GKB < K) {  // gfx1250 global_prefetch hint for next K-tile
      __builtin_prefetch(Ab + (size_t)(tid >> 1) * K + k0 + GKB, 0, 1);
    }
    async_wait0();
    __syncthreads();
    v16bf a0 = load_frag_a(sA + (wm + 0) * GLD, GLD, lane);
    v16bf a1 = load_frag_a(sA + (wm + 16) * GLD, GLD, lane);
    v16bf b0 = load_frag_b(sB + (wn + 0) * GLD, GLD, lane);
    v16bf b1 = load_frag_b(sB + (wn + 16) * GLD, GLD, lane);
    acc[0][0] = wmma_bf16(a0, b0, acc[0][0]);
    acc[0][1] = wmma_bf16(a0, b1, acc[0][1]);
    acc[1][0] = wmma_bf16(a1, b0, acc[1][0]);
    acc[1][1] = wmma_bf16(a1, b1, acc[1][1]);
    __syncthreads();
  }
  int half = lane >> 4, idx = lane & 15;
  #pragma unroll
  for (int fm = 0; fm < 2; ++fm)
    #pragma unroll
    for (int fn = 0; fn < 2; ++fn) {
      int n = bn + wn + fn * 16 + idx;
      int mb = bm + wm + fm * 16 + half * 8;
      float bia = (flags & GF_BIAS) ? bias[n] : 0.f;
      #pragma unroll
      for (int e = 0; e < 8; ++e) {
        size_t o = (size_t)(mb + e) * N + n;
        float v = acc[fm][fn][e] + bia;
        if (flags & GF_ACCUM) v += Cf[o];
        if (flags & GF_RELU) v = fmaxf(v, 0.f);
        if (Cf) Cf[o] = v;
        if (Cbf) Cbf[o] = f2bf(v);
      }
    }
}

// ---------------------------------------------------------------------------
// QKV GEMM: [4096,512] x [1536,512]^T + bias; epilogue scatters to
//   q,k : [B,H,S,64] bf16 (row-major per head)   vT : [B,H,64,S] bf16
// ---------------------------------------------------------------------------
__global__ __launch_bounds__(256)
void qkv_gemm_kernel(const bf16_t* __restrict__ A, const bf16_t* __restrict__ Bw,
                     const float* __restrict__ bias, bf16_t* __restrict__ qo,
                     bf16_t* __restrict__ ko, bf16_t* __restrict__ vto, int K) {
  __shared__ __align__(16) bf16_t sA[GBM * GLD];
  __shared__ __align__(16) bf16_t sB[GBN * GLD];
  int tid = threadIdx.x, lane = tid & 31, wv = tid >> 5;
  int wm = (wv >> 1) * 32, wn = (wv & 1) * 32;
  int bm = blockIdx.y * GBM, bn = blockIdx.x * GBN;
  const bf16_t* Ab = A + (size_t)bm * K;
  const bf16_t* Bb = Bw + (size_t)bn * K;
  v8f acc[2][2] = {};
  for (int k0 = 0; k0 < K; k0 += GKB) {
    for (int u = tid; u < GBM * 4; u += 256) {
      int r = u >> 2, s4 = u & 3;
      async_copy_b128(sA + r * GLD + s4 * 8, Ab + (size_t)r * K + k0 + s4 * 8);
    }
    for (int u = tid; u < GBN * 4; u += 256) {
      int r = u >> 2, s4 = u & 3;
      async_copy_b128(sB + r * GLD + s4 * 8, Bb + (size_t)r * K + k0 + s4 * 8);
    }
    async_wait0();
    __syncthreads();
    v16bf a0 = load_frag_a(sA + (wm + 0) * GLD, GLD, lane);
    v16bf a1 = load_frag_a(sA + (wm + 16) * GLD, GLD, lane);
    v16bf b0 = load_frag_b(sB + (wn + 0) * GLD, GLD, lane);
    v16bf b1 = load_frag_b(sB + (wn + 16) * GLD, GLD, lane);
    acc[0][0] = wmma_bf16(a0, b0, acc[0][0]);
    acc[0][1] = wmma_bf16(a0, b1, acc[0][1]);
    acc[1][0] = wmma_bf16(a1, b0, acc[1][0]);
    acc[1][1] = wmma_bf16(a1, b1, acc[1][1]);
    __syncthreads();
  }
  int half = lane >> 4, idx = lane & 15;
  #pragma unroll
  for (int fm = 0; fm < 2; ++fm)
    #pragma unroll
    for (int fn = 0; fn < 2; ++fn) {
      int n = bn + wn + fn * 16 + idx;
      int mb = bm + wm + fm * 16 + half * 8;
      float bia = bias[n];
      #pragma unroll
      for (int e = 0; e < 8; ++e) {
        int m = mb + e, b = m >> 10, s = m & 1023;
        bf16_t bv = f2bf(acc[fm][fn][e] + bia);
        if (n < 512) {
          int h = n >> 6, d = n & 63;
          qo[((size_t)(b * 8 + h) * 1024 + s) * 64 + d] = bv;
        } else if (n < 1024) {
          int n2 = n - 512, h = n2 >> 6, d = n2 & 63;
          ko[((size_t)(b * 8 + h) * 1024 + s) * 64 + d] = bv;
        } else {
          int n2 = n - 1024, h = n2 >> 6, d = n2 & 63;
          vto[((size_t)(b * 8 + h) * 64 + d) * 1024 + s] = bv;
        }
      }
    }
}

// ---------------------------------------------------------------------------
// Fused attention: per workgroup = (b, 16-row stripe). For each head:
//   scores (WMMA) -> LDS fp32 stripe -> softmax in-place -> attn_avg accumulate
//   -> ctx = P.V^T (WMMA, K split across waves, ds_add_f32 reduce) -> ctx_bf.
// ---------------------------------------------------------------------------
#define SSTR 1044   // 1024 + 20-float pad: 16B aligned rows, conflict-free frag reads

__global__ __launch_bounds__(256)
void attn_kernel(const bf16_t* __restrict__ qb, const bf16_t* __restrict__ kb,
                 const bf16_t* __restrict__ vtb, float* __restrict__ attn_avg,
                 bf16_t* __restrict__ ctx) {
  __shared__ __align__(16) float sS[16 * SSTR];
  __shared__ float sctx[16 * 64];
  int tid = threadIdx.x, lane = tid & 31, wv = tid >> 5;
  int half = lane >> 4, idx = lane & 15;
  int b = blockIdx.x >> 6;
  int i0 = (blockIdx.x & 63) << 4;

  for (int h = 0; h < 8; ++h) {
    const bf16_t* qbase = qb + ((size_t)(b * 8 + h) * 1024 + i0) * 64;
    const bf16_t* kbase = kb + (size_t)(b * 8 + h) * 1024 * 64;
    const bf16_t* vbase = vtb + (size_t)(b * 8 + h) * 64 * 1024;

    // ---- scores: 16 x 1024, each wave owns 128 columns ----
    v16bf aq0 = load_frag_a(qbase, 64, lane);
    v16bf aq1 = load_frag_a(qbase + 32, 64, lane);
    for (int nt = 0; nt < 8; ++nt) {
      int n0 = wv * 128 + nt * 16;
      v8f sc = {};
      sc = wmma_bf16(aq0, load_frag_b(kbase + (size_t)n0 * 64, 64, lane), sc);
      sc = wmma_bf16(aq1, load_frag_b(kbase + (size_t)n0 * 64 + 32, 64, lane), sc);
      #pragma unroll
      for (int e = 0; e < 8; ++e)
        sS[(half * 8 + e) * SSTR + n0 + idx] = sc[e] * 0.125f;  // 1/sqrt(64)
    }
    __syncthreads();

    // ---- softmax (wave wv owns rows 2wv,2wv+1), accumulate head-average ----
    #pragma unroll
    for (int rr = 0; rr < 2; ++rr) {
      int r = wv * 2 + rr;
      float* row = sS + r * SSTR;
      float m = -1e30f;
      #pragma unroll
      for (int t = 0; t < 32; ++t) m = fmaxf(m, row[lane + 32 * t]);
      m = wred_max(m);
      float pv[32], s = 0.f;
      #pragma unroll
      for (int t = 0; t < 32; ++t) { pv[t] = __expf(row[lane + 32 * t] - m); s += pv[t]; }
      s = wred_sum(s);
      float inv = 1.f / s;
      float* ga = attn_avg + ((size_t)b * 1024 + i0 + r) * 1024;
      #pragma unroll
      for (int t = 0; t < 32; ++t) {
        int j = lane + 32 * t;
        float p = pv[t] * inv;
        row[j] = p;
        if (h == 0) ga[j] = p * 0.125f;       // mean over H=8 heads
        else        ga[j] += p * 0.125f;
      }
    }
    for (int u = tid; u < 16 * 64; u += 256) sctx[u] = 0.f;
    __syncthreads();

    // ---- ctx = P[16,1024] . V^T[64,1024]^T ; wave wv owns K in [128wv,128wv+128) ----
    {
      v8f ca[4] = {};
      for (int ks = 0; ks < 4; ++ks) {
        int kg = wv * 128 + ks * 32;
        const float* arow = sS + idx * SSTR + kg;
        v16bf av;
        #pragma unroll
        for (int t = 0; t < 8; ++t) av[t] = f2bf(arow[8 * half + t]);
        #pragma unroll
        for (int t = 0; t < 8; ++t) av[8 + t] = f2bf(arow[16 + 8 * half + t]);
        #pragma unroll
        for (int nt = 0; nt < 4; ++nt) {
          v16bf bv = load_frag_b(vbase + (size_t)(nt * 16) * 1024 + kg, 1024, lane);
          ca[nt] = wmma_bf16(av, bv, ca[nt]);
        }
      }
      #pragma unroll
      for (int nt = 0; nt < 4; ++nt) {
        int n = nt * 16 + idx;
        #pragma unroll
        for (int e = 0; e < 8; ++e)
          atomicAdd(&sctx[(half * 8 + e) * 64 + n], ca[nt][e]);  // ds_add_f32
      }
    }
    __syncthreads();
    for (int u = tid; u < 16 * 64; u += 256) {
      int r = u >> 6, c2 = u & 63;
      ctx[((size_t)(b * 1024 + i0 + r)) * 512 + h * 64 + c2] = f2bf(sctx[u]);
    }
    __syncthreads();
  }
}

// ---------------------------------------------------------------------------
// Edge branch. softmax(w_c*A + b_c) = softmax(w_c*A); result = w_c*E_p[A]+b_c.
// One wave per row of attn_avg; row cached in registers, C=32 passes.
// ---------------------------------------------------------------------------
__global__ __launch_bounds__(256)
void edge_kernel(const float* __restrict__ avg, const float* __restrict__ ew,
                 const float* __restrict__ eb, float* __restrict__ wsum) {
  int wv = threadIdx.x >> 5, lane = threadIdx.x & 31;
  int row = blockIdx.x * 8 + wv;
  const float* a = avg + (size_t)row * 1024;
  float va[32];
  #pragma unroll
  for (int t = 0; t < 32; ++t) va[t] = a[lane + 32 * t];
  for (int c = 0; c < 32; ++c) {
    float wc = ew[c];
    float m = -1e30f;
    #pragma unroll
    for (int t = 0; t < 32; ++t) m = fmaxf(m, wc * va[t]);
    m = wred_max(m);
    float s = 0.f, sa = 0.f;
    #pragma unroll
    for (int t = 0; t < 32; ++t) {
      float e = __expf(wc * va[t] - m);
      s += e; sa += e * va[t];
    }
    s = wred_sum(s); sa = wred_sum(sa);
    if (lane == 0) wsum[(size_t)row * 32 + c] = wc * (sa / s) + eb[c];
  }
}

// ---------------------------------------------------------------------------
// LayerNorm(x1 + x2) * g + b ; one wave per D=512 row.
// ---------------------------------------------------------------------------
__global__ __launch_bounds__(256)
void ln_kernel(const float* __restrict__ x1, const float* __restrict__ x2,
               const float* __restrict__ g, const float* __restrict__ bt,
               float* __restrict__ outf, bf16_t* __restrict__ outbf) {
  int wv = threadIdx.x >> 5, lane = threadIdx.x & 31;
  int row = blockIdx.x * 8 + wv;
  const float* p1 = x1 + (size_t)row * 512;
  const float* p2 = x2 + (size_t)row * 512;
  float v[16], s = 0.f;
  #pragma unroll
  for (int t = 0; t < 16; ++t) { v[t] = p1[lane + 32 * t] + p2[lane + 32 * t]; s += v[t]; }
  float mean = wred_sum(s) * (1.f / 512.f);
  float q = 0.f;
  #pragma unroll
  for (int t = 0; t < 16; ++t) { float d = v[t] - mean; q += d * d; }
  float inv = rsqrtf(wred_sum(q) * (1.f / 512.f) + 1e-5f);
  #pragma unroll
  for (int t = 0; t < 16; ++t) {
    int d = lane + 32 * t;
    float y = (v[t] - mean) * inv * g[d] + bt[d];
    if (outf) outf[(size_t)row * 512 + d] = y;
    if (outbf) outbf[(size_t)row * 512 + d] = f2bf(y);
  }
}

// ---------------------------------------------------------------------------
// Host launcher
// ---------------------------------------------------------------------------
extern "C" void kernel_launch(void* const* d_in, const int* in_sizes, int n_in,
                              void* d_out, int out_size, void* d_ws, size_t ws_size,
                              hipStream_t stream) {
  (void)in_sizes; (void)n_in; (void)out_size; (void)ws_size;
  const float* src        = (const float*)d_in[0];
  const float* in_proj_w  = (const float*)d_in[1];
  const float* in_proj_b  = (const float*)d_in[2];
  const float* out_proj_w = (const float*)d_in[3];
  const float* out_proj_b = (const float*)d_in[4];
  const float* expand_w   = (const float*)d_in[5];
  const float* expand_b   = (const float*)d_in[6];
  const float* proj_w     = (const float*)d_in[7];
  const float* proj_b     = (const float*)d_in[8];
  const float* lin1_w     = (const float*)d_in[9];
  const float* lin1_b     = (const float*)d_in[10];
  const float* lin2_w     = (const float*)d_in[11];
  const float* lin2_b     = (const float*)d_in[12];
  const float* ln1_g      = (const float*)d_in[13];
  const float* ln1_bb     = (const float*)d_in[14];
  const float* ln2_g      = (const float*)d_in[15];
  const float* ln2_bb     = (const float*)d_in[16];

  char* ws = (char*)d_ws;
  size_t off = 0;
  auto take = [&](size_t bytes) -> char* {
    size_t o = off;
    off = (off + bytes + 255) & ~(size_t)255;
    return ws + o;
  };
  const int BS = 4096, D = 512, FF = 2048, C = 32;
  bf16_t* srcbf  = (bf16_t*)take((size_t)BS * D * 2);
  bf16_t* w_in   = (bf16_t*)take((size_t)3 * D * D * 2);
  bf16_t* w_out  = (bf16_t*)take((size_t)D * D * 2);
  bf16_t* w_l1   = (bf16_t*)take((size_t)FF * D * 2);
  bf16_t* w_l2   = (bf16_t*)take((size_t)D * FF * 2);
  bf16_t* w_pr   = (bf16_t*)take((size_t)D * C * 2);
  bf16_t* qbf    = (bf16_t*)take((size_t)BS * D * 2);
  bf16_t* kbf    = (bf16_t*)take((size_t)BS * D * 2);
  bf16_t* vtbf   = (bf16_t*)take((size_t)BS * D * 2);
  bf16_t* ctxbf  = (bf16_t*)take((size_t)BS * D * 2);
  bf16_t* xbf    = (bf16_t*)take((size_t)BS * D * 2);
  bf16_t* ff1    = (bf16_t*)take((size_t)BS * FF * 2);
  bf16_t* wsumbf = (bf16_t*)take((size_t)BS * C * 2);
  float*  avg    = (float*)take((size_t)4 * 1024 * 1024 * 4);
  float*  src2   = (float*)take((size_t)BS * D * 4);
  float*  xf     = (float*)take((size_t)BS * D * 4);
  float*  ffout  = (float*)take((size_t)BS * D * 4);
  float*  wsum   = (float*)take((size_t)BS * C * 4);

  auto cvt = [&](const float* in, bf16_t* o, int n) {
    int blocks = (n + 1023) / 1024; if (blocks > 2048) blocks = 2048;
    cvt_kernel<<<blocks, 256, 0, stream>>>(in, o, n);
  };
  cvt(src, srcbf, BS * D);
  cvt(in_proj_w, w_in, 3 * D * D);
  cvt(out_proj_w, w_out, D * D);
  cvt(lin1_w, w_l1, FF * D);
  cvt(lin2_w, w_l2, D * FF);
  cvt(proj_w, w_pr, D * C);

  // QKV projection + head-layout scatter
  qkv_gemm_kernel<<<dim3((3 * D) / GBN, BS / GBM), 256, 0, stream>>>(
      srcbf, w_in, in_proj_b, qbf, kbf, vtbf, D);

  // Fused attention (scores/softmax/ctx + head-averaged attention)
  attn_kernel<<<256, 256, 0, stream>>>(qbf, kbf, vtbf, avg, ctxbf);

  // out-proj -> src2
  gemm_bf16_kernel<<<dim3(D / GBN, BS / GBM), 256, 0, stream>>>(
      ctxbf, w_out, out_proj_b, src2, nullptr, BS, D, D, GF_BIAS);

  // edge branch -> wsum[B*S, C], then project into src2
  edge_kernel<<<512, 256, 0, stream>>>(avg, expand_w, expand_b, wsum);
  cvt(wsum, wsumbf, BS * C);
  gemm_bf16_kernel<<<dim3(D / GBN, BS / GBM), 256, 0, stream>>>(
      wsumbf, w_pr, proj_b, src2, nullptr, BS, D, C, GF_BIAS | GF_ACCUM);

  // x = LN1(src + src2)
  ln_kernel<<<512, 256, 0, stream>>>(src, src2, ln1_g, ln1_bb, xf, xbf);

  // FFN
  gemm_bf16_kernel<<<dim3(FF / GBN, BS / GBM), 256, 0, stream>>>(
      xbf, w_l1, lin1_b, nullptr, ff1, BS, FF, D, GF_BIAS | GF_RELU);
  gemm_bf16_kernel<<<dim3(D / GBN, BS / GBM), 256, 0, stream>>>(
      ff1, w_l2, lin2_b, ffout, nullptr, BS, D, FF, GF_BIAS);

  // out = LN2(x + ff)
  ln_kernel<<<512, 256, 0, stream>>>(xf, ffout, ln2_g, ln2_bb, (float*)d_out, nullptr);
}
